// CrystalAE_86071144612483
// MI455X (gfx1250) — matrix operand order repaired and checked by reading, not compiled, not run on hardware
//
#include <hip/hip_runtime.h>
#include <hip/hip_bf16.h>

typedef __attribute__((ext_vector_type(16))) __bf16 v16bf;
typedef __attribute__((ext_vector_type(8)))  float  v8f;

#define N_ATOMS 32768
#define M_NBR   12
#define FEA     64
#define NBRF    41
#define EPAD    44      // padded edge-feature row (zeros at 41..43)
#define CIN     169
#define KPAD    192
#define COUT    128
#define ATOMS_PER_BLK 16
#define NTILES  12
#define HCOLS   480     // 6*64 bilinear cols + 96 (padded 92) atom-out cols

__device__ __forceinline__ float splus(float x){
  return (x > 20.f) ? x : log1pf(expf(x));
}

// ---------------------------------------------------------------------------
// Conv GEMM: A = concat(self_h, nbr_h, e) [16 atoms * 12 nbrs = 192 rows x 192K]
//            B = conv weights bf16 [128 cols x 192 K], C = z [192 x 128]
// PASS 0: accumulate per-channel sum / sumsq of z (for BN1 batch stats)
// PASS 1: apply BN1, sigmoid(gate)*softplus(core), sum over M -> s[N,64]
// ---------------------------------------------------------------------------
template<int PASS>
__global__ __launch_bounds__(256)
void k_conv_gemm(const __bf16* __restrict__ hbf,     // [N,64]
                 const __bf16* __restrict__ ebf,     // [N,12,44] (padded)
                 const int*    __restrict__ nbr_idx, // [N,12]
                 const __bf16* __restrict__ Wt,      // [128,192]
                 const float*  __restrict__ bias,    // [128]
                 const float*  __restrict__ bn1_ss,  // [256]: scale, shift
                 float*        __restrict__ part,    // [grid,256] partial stats
                 float*        __restrict__ s_out)   // [N,64]
{
  __shared__ __align__(32) __bf16 aS[6*32*16];  // A fragments, one 16x192 tile
  __shared__ float zt[16*128];                  // BN'd z tile (PASS 1)
  __shared__ float sacc[ATOMS_PER_BLK*64];      // per-atom gated sums (PASS 1)
  __shared__ float red[512];                    // stats reduction (PASS 0)

  const int tid  = threadIdx.x;
  const int wave = tid >> 5, lane = tid & 31;
  const int ln   = lane & 15, hi = lane >> 4;
  const int col  = wave*16 + ln;                // output channel owned by lane
  const int n0   = blockIdx.x * ATOMS_PER_BLK;

  // B fragments: 32 contiguous bytes of [col][K] layout per fragment
  v16bf bfr[6];
  {
    const __bf16* wc = Wt + col*KPAD + 16*hi;
    #pragma unroll
    for (int kc=0; kc<6; ++kc)
      bfr[kc] = *((const v16bf*)(wc + kc*32));
  }
  const float bcol = bias[col];
  float sc = 0.f, sh = 0.f;
  if (PASS == 1){ sc = bn1_ss[col]; sh = bn1_ss[128+col]; }

  // fixed staging slot: thread owns (lane_s, vgpr pair vv)
  const int lane_s = tid >> 3;                  // 0..31
  const int vv     = tid & 7;
  const int hi_s   = lane_s >> 4;
  const int mrow   = lane_s & 15;
  const int kbase  = (hi_s?16:0) + (vv<<1);
  const unsigned* h32 = (const unsigned*)hbf;
  const unsigned* e32 = (const unsigned*)ebf;
  unsigned* aS32 = (unsigned*)aS;
  const int slot = lane_s*8 + vv;

  float csum = 0.f, csq = 0.f;
  if (PASS == 1){
    for (int i=tid; i<ATOMS_PER_BLK*64; i+=256) sacc[i] = 0.f;
  }
  __syncthreads();

  for (int t=0; t<NTILES; ++t){
    { // stage A tile: row decode once, 6 u32 pair copies
      int rr = t*16 + mrow;
      int a  = rr/12;
      int mm = rr - a*12;
      int n  = n0 + a;
      int nb = nbr_idx[n*M_NBR + mm];
      int er = (n*M_NBR + mm)*(EPAD/2);
      #pragma unroll
      for (int kc=0; kc<6; ++kc){
        int k = kc*32 + kbase;
        unsigned val;
        if      (k < FEA)        val = h32[n*(FEA/2)  + (k>>1)];
        else if (k < 2*FEA)      val = h32[nb*(FEA/2) + ((k-FEA)>>1)];
        else if (k < 2*FEA+EPAD) val = e32[er + ((k-2*FEA)>>1)];
        else                     val = 0u;
        aS32[kc*256 + slot] = val;
      }
    }
    __syncthreads();

    v16bf afr[6];
    #pragma unroll
    for (int kc=0; kc<6; ++kc)
      afr[kc] = *((const v16bf*)(aS + (kc*32 + lane)*16));

    v8f acc;
    #pragma unroll
    for (int v=0; v<8; ++v) acc[v] = bcol;      // z includes conv bias
    #pragma unroll
    for (int kc=0; kc<6; ++kc)
      acc = __builtin_amdgcn_wmma_f32_16x16x32_bf16(
          false, afr[kc], false, bfr[kc], (short)0, acc, false, false);

    if (PASS == 0){
      #pragma unroll
      for (int v=0; v<8; ++v){ float z = acc[v]; csum += z; csq += z*z; }
      __syncthreads();
    } else {
      #pragma unroll
      for (int v=0; v<8; ++v)
        zt[(v + hi*8)*128 + col] = acc[v]*sc + sh;  // C row M = v + 8*hi
      __syncthreads();
      if (tid < 128){
        int a_rel = tid >> 6, c = tid & 63;
        int a_l   = (16*t)/12 + a_rel;          // 16 rows span <=2 atoms
        if (a_l < ATOMS_PER_BLK){
          float pacc = 0.f;
          #pragma unroll
          for (int m=0; m<16; ++m){
            int rr = 16*t + m;
            if (rr/12 == a_l){
              float g = zt[m*128 + c];
              float u = zt[m*128 + 64 + c];
              pacc += (1.f/(1.f+expf(-g))) * splus(u);
            }
          }
          sacc[a_l*64 + c] += pacc;
        }
      }
      __syncthreads();
    }
  }

  if (PASS == 0){
    red[tid] = csum; red[256+tid] = csq;
    __syncthreads();
    if (hi == 0){
      part[blockIdx.x*256 + col]       = red[tid]     + red[tid+16];
      part[blockIdx.x*256 + 128 + col] = red[256+tid] + red[256+tid+16];
    }
  } else {
    for (int o=tid; o<ATOMS_PER_BLK*64; o+=256)
      s_out[(n0 + (o>>6))*FEA + (o&63)] = sacc[o];
  }
}

// ---------------------------------------------------------------------------
// Head GEMM: U = H[16,64] @ Wcat[64,480]  (cols 0..383 = bilinear W_k,
//            cols 384..475 = fcA cols; 476..479 zero pad)
// Epilogue: y[n,k] = sum_e U[n,k*64+e]*h[n,e] + bilin_b[k]; fc1; log_softmax.
// atom_out tiles (+bias) go straight to global.
// ---------------------------------------------------------------------------
__global__ __launch_bounds__(256)
void k_headout(const __bf16* __restrict__ hbf, const float* __restrict__ hf,
               const __bf16* __restrict__ Wcat,   // [480][64]
               const float* __restrict__ bilb, const float* __restrict__ f1W,
               const float* __restrict__ f1b,  const float* __restrict__ fcAb,
               float* __restrict__ ybuf, float* __restrict__ aout)
{
  __shared__ __align__(32) __bf16 aS[2*32*16];  // H fragments (2 k-chunks)
  __shared__ float U[16*384];                   // bilinear GEMM result
  __shared__ float yk[16*8];                    // pre-softmax y per atom

  const int tid = threadIdx.x;
  const int wave = tid >> 5, lane = tid & 31;
  const int ln = lane & 15, hi = lane >> 4;
  const int n0 = blockIdx.x * 16;

  { // stage A = H rows n0..n0+15, K = 0..63
    const int lane_s = tid >> 3, vv = tid & 7;
    const int hi_s = lane_s >> 4, mrow = lane_s & 15;
    const int kbase = (hi_s?16:0) + (vv<<1);
    const unsigned* h32 = (const unsigned*)hbf;
    unsigned* aS32 = (unsigned*)aS;
    int n = n0 + mrow;
    #pragma unroll
    for (int kc=0; kc<2; ++kc)
      aS32[kc*256 + lane_s*8 + vv] = h32[n*(FEA/2) + ((kc*32 + kbase)>>1)];
  }
  __syncthreads();
  v16bf a0 = *((const v16bf*)(aS + (0*32 + lane)*16));
  v16bf a1 = *((const v16bf*)(aS + (1*32 + lane)*16));

  for (int ti = wave; ti < 30; ti += 8){        // 24 bilinear + 6 atom-out tiles
    int colbase = ti*16;
    const __bf16* wc = Wcat + (colbase + ln)*FEA + 16*hi;
    v16bf b0 = *((const v16bf*)(wc));
    v16bf b1 = *((const v16bf*)(wc + 32));
    v8f acc;
    #pragma unroll
    for (int v=0; v<8; ++v) acc[v] = 0.f;
    acc = __builtin_amdgcn_wmma_f32_16x16x32_bf16(false, a0, false, b0, (short)0, acc, false, false);
    acc = __builtin_amdgcn_wmma_f32_16x16x32_bf16(false, a1, false, b1, (short)0, acc, false, false);
    if (ti < 24){
      #pragma unroll
      for (int v=0; v<8; ++v)
        U[(v + 8*hi)*384 + colbase + ln] = acc[v];
    } else {
      int o = colbase - 384 + ln;               // 0..95
      if (o < 92){
        float bb = fcAb[o];
        #pragma unroll
        for (int v=0; v<8; ++v)
          aout[(n0 + v + 8*hi)*92 + o] = acc[v] + bb;
      }
    }
  }
  __syncthreads();

  if (tid < 96){                                // 16 atoms x 6 k row-dots
    int na = tid & 15, k = tid >> 4;
    const float* hn = hf + (n0 + na)*FEA;
    const float* un = U + na*384 + k*64;
    float acc = bilb[k];
    #pragma unroll 8
    for (int e=0; e<64; ++e) acc += un[e]*hn[e];
    yk[na*8 + k] = acc;
  }
  __syncthreads();

  if (tid < 16){                                // fc1 + log_softmax per atom
    float z[6], mx = -1e30f;
    #pragma unroll
    for (int k=0; k<6; ++k){
      float a = f1b[k];
      #pragma unroll
      for (int j=0; j<6; ++j) a += f1W[k*6+j]*yk[tid*8+j];
      z[k] = a; mx = fmaxf(mx, a);
    }
    float se = 0.f;
    #pragma unroll
    for (int k=0; k<6; ++k) se += expf(z[k]-mx);
    float ls = mx + logf(se);
    #pragma unroll
    for (int k=0; k<6; ++k) ybuf[(n0+tid)*6 + k] = z[k]-ls;
  }
}

// --- weight conversion: conv_W [3,169,128] f32 -> Wt [3,128,192] bf16 ---
__global__ void k_prep_w(const float* __restrict__ convW, __bf16* __restrict__ Wt){
  int t = blockIdx.x*256 + threadIdx.x;       // 3*128*192
  int l = t / (COUT*KPAD);
  int r = t - l*(COUT*KPAD);
  int c = r / KPAD;
  int k = r - c*KPAD;
  float v = (k < CIN) ? convW[(l*CIN + k)*COUT + c] : 0.f;
  Wt[t] = (__bf16)v;
}

// --- head weights: Wcat [480][64] bf16 = [bilin_W cols | fcA rows | pad] ---
__global__ void k_prep_wcat(const float* __restrict__ bilW,
                            const float* __restrict__ fcAW,
                            __bf16* __restrict__ Wcat){
  int t = blockIdx.x*256 + threadIdx.x;       // 480*64
  int c = t >> 6, d = t & 63;
  float v;
  if (c < 384){ int k = c >> 6, e = c & 63; v = bilW[k*4096 + d*64 + e]; }
  else        { int o = c - 384; v = (o < 92) ? fcAW[o*64 + d] : 0.f; }
  Wcat[t] = (__bf16)v;
}

// --- node embedding: h = ari_table[atom_fea] @ emb_W -> f32 + bf16 copies ---
__global__ void k_embed(const int* __restrict__ atom_fea, const float* __restrict__ ari,
                        const float* __restrict__ embW, float* __restrict__ hf,
                        __bf16* __restrict__ hb){
  __shared__ float xs[4*92];
  int a0 = blockIdx.x*4;
  for (int i=threadIdx.x; i<4*92; i+=256){
    int a = i/92, d = i - a*92;
    xs[i] = ari[atom_fea[a0+a]*92 + d];
  }
  __syncthreads();
  int a = threadIdx.x >> 6, c = threadIdx.x & 63;
  const float* x = xs + a*92;
  float acc = 0.f;
  #pragma unroll 4
  for (int d=0; d<92; ++d) acc += x[d]*embW[d*FEA + c];
  int idx = (a0+a)*FEA + c;
  hf[idx] = acc; hb[idx] = (__bf16)acc;
}

// --- Gaussian edge expansion -> bf16, rows padded 41 -> 44 with zeros ---
__global__ void k_edge(const float* __restrict__ nbr_fea, __bf16* __restrict__ eb){
  int i  = blockIdx.x*256 + threadIdx.x;      // N*M*44
  int nm = i / EPAD;
  int f  = i - nm*EPAD;
  float v = 0.f;
  if (f < NBRF){
    float d = nbr_fea[nm] - 0.2f*(float)f;
    v = expf(-25.f*d*d);
  }
  eb[i] = (__bf16)v;
}

// --- finalize BN1 batch stats -> scale/shift per channel ---
__global__ void k_bn1_fin(const float* __restrict__ part, int nblk,
                          const float* __restrict__ g, const float* __restrict__ b,
                          float* __restrict__ ss){
  int c = threadIdx.x;                        // 128
  float sum = 0.f, sq = 0.f;
  for (int i=0; i<nblk; ++i){ sum += part[i*256+c]; sq += part[i*256+128+c]; }
  const float inv = 1.f/(float)(N_ATOMS*M_NBR);
  float m   = sum*inv;
  float var = sq*inv - m*m;
  float scale = g[c]*rsqrtf(var + 1e-5f);
  ss[c] = scale; ss[128+c] = b[c] - m*scale;
}

// --- BN2 batch stats over s [N,64] -> scale/shift ---
__global__ void k_bn2_stats(const float* __restrict__ s, const float* __restrict__ g,
                            const float* __restrict__ b, float* __restrict__ ss){
  __shared__ float r1[256], r2[256];
  int c = blockIdx.x;                         // 64
  float sum = 0.f, sq = 0.f;
  for (int n=threadIdx.x; n<N_ATOMS; n+=256){ float v = s[n*FEA+c]; sum+=v; sq+=v*v; }
  r1[threadIdx.x] = sum; r2[threadIdx.x] = sq;
  __syncthreads();
  for (int off=128; off; off>>=1){
    if (threadIdx.x < off){
      r1[threadIdx.x] += r1[threadIdx.x+off];
      r2[threadIdx.x] += r2[threadIdx.x+off];
    }
    __syncthreads();
  }
  if (threadIdx.x == 0){
    const float inv = 1.f/(float)N_ATOMS;
    float m   = r1[0]*inv;
    float var = r2[0]*inv - m*m;
    float scale = g[c]*rsqrtf(var + 1e-5f);
    ss[c] = scale; ss[64+c] = b[c] - m*scale;
  }
}

// --- h = softplus(h + BN2(s)); refresh f32 + bf16 copies ---
__global__ void k_update_h(const float* __restrict__ s, const float* __restrict__ ss,
                           float* __restrict__ hf, __bf16* __restrict__ hb){
  int i = blockIdx.x*256 + threadIdx.x;       // N*64
  int c = i & 63;
  float x = hf[i] + s[i]*ss[c] + ss[64+c];
  float o = splus(x);
  hf[i] = o; hb[i] = (__bf16)o;
}

// --- adj output: adj[c,i,j,k] = y[cai[c,j], k]  (pure streaming write) ---
__global__ void k_adj(const float* __restrict__ y, const int* __restrict__ cai,
                      float* __restrict__ out){
  int idx  = blockIdx.x*256 + threadIdx.x;
  int row  = idx/6;
  int k    = idx - row*6;
  int c    = row >> 14;                       // /(128*128)
  int j    = row & 127;
  int atom = cai[c*128 + j];
  out[idx] = y[atom*6 + k];
}

extern "C" void kernel_launch(void* const* d_in, const int* in_sizes, int n_in,
                              void* d_out, int out_size, void* d_ws, size_t ws_size,
                              hipStream_t stream){
  const int*   atom_fea = (const int*)  d_in[0];
  const float* nbr_fea  = (const float*)d_in[1];
  const int*   nbr_idx  = (const int*)  d_in[2];
  const int*   cai      = (const int*)  d_in[3];
  const float* ari      = (const float*)d_in[4];
  const float* embW     = (const float*)d_in[5];
  const float* convW    = (const float*)d_in[6];
  const float* convb    = (const float*)d_in[7];
  const float* bn1g     = (const float*)d_in[8];
  const float* bn1b     = (const float*)d_in[9];
  const float* bn2g     = (const float*)d_in[10];
  const float* bn2b     = (const float*)d_in[11];
  const float* bilW     = (const float*)d_in[12];
  const float* bilb     = (const float*)d_in[13];
  const float* f1W      = (const float*)d_in[14];
  const float* f1b      = (const float*)d_in[15];
  const float* fcAW     = (const float*)d_in[16];
  const float* fcAb     = (const float*)d_in[17];
  float* out = (float*)d_out;

  // workspace carve-up (~60 MB, all L2-resident intermediates)
  char* ws = (char*)d_ws;
  size_t off = 0;
  auto carve = [&](size_t bytes)->char*{
    char* p = ws + off; off = (off + bytes + 255) & ~size_t(255); return p;
  };
  float*  hf   = (float*) carve(size_t(N_ATOMS)*FEA*4);
  __bf16* hb   = (__bf16*)carve(size_t(N_ATOMS)*FEA*2);
  __bf16* eb   = (__bf16*)carve(size_t(N_ATOMS)*M_NBR*EPAD*2);
  __bf16* Wt   = (__bf16*)carve(size_t(3)*COUT*KPAD*2);
  __bf16* Wcat = (__bf16*)carve(size_t(HCOLS)*FEA*2);
  float*  part = (float*) carve(size_t(2048)*256*4);
  float*  ss1  = (float*) carve(256*4);
  float*  ss2  = (float*) carve(128*4);
  float*  sbuf = (float*) carve(size_t(N_ATOMS)*FEA*4);
  float*  ybuf = (float*) carve(size_t(N_ATOMS)*6*4);

  k_prep_w   <<<288, 256, 0, stream>>>(convW, Wt);
  k_prep_wcat<<<(HCOLS*FEA)/256, 256, 0, stream>>>(bilW, fcAW, Wcat);
  k_embed    <<<N_ATOMS/4, 256, 0, stream>>>(atom_fea, ari, embW, hf, hb);
  k_edge     <<<(N_ATOMS*M_NBR*EPAD)/256, 256, 0, stream>>>(nbr_fea, eb);

  const int gblk = N_ATOMS/ATOMS_PER_BLK;     // 2048
  for (int l=0; l<3; ++l){
    const __bf16* Wl = Wt + (size_t)l*COUT*KPAD;
    const float*  bl = convb + l*COUT;
    k_conv_gemm<0><<<gblk, 256, 0, stream>>>(hb, eb, nbr_idx, Wl, bl, ss1, part, sbuf);
    k_bn1_fin<<<1, 128, 0, stream>>>(part, gblk, bn1g + l*COUT, bn1b + l*COUT, ss1);
    k_conv_gemm<1><<<gblk, 256, 0, stream>>>(hb, eb, nbr_idx, Wl, bl, ss1, part, sbuf);
    k_bn2_stats<<<64, 256, 0, stream>>>(sbuf, bn2g + l*FEA, bn2b + l*FEA, ss2);
    k_update_h<<<(N_ATOMS*FEA)/256, 256, 0, stream>>>(sbuf, ss2, hf, hb);
  }

  k_headout<<<gblk, 256, 0, stream>>>(hb, hf, Wcat, bilb, f1W, f1b, fcAb,
                                      ybuf, out + (size_t)256*128*128*6);
  k_adj<<<(256*128*128*6)/256, 256, 0, stream>>>(ybuf, cai, out);
}
